// TransientGenerator_56118042689688
// MI455X (gfx1250) — matrix operand order repaired and checked by reading, not compiled, not run on hardware
//
#include <hip/hip_runtime.h>
#include <hip/hip_bf16.h>

// Problem constants (match reference)
#define HOPS   240
#define NATOM  16
#define ALEN   32
#define NB     32
#define NC     128
#define NT     4096
#define AUDIO_LEN ((NT - 1) * HOPS + ALEN)   // 982832

// Tile geometry for the fused front-end
#define LW    48     // LDS tile width (time positions incl. halo) = 3 WMMA N-tiles
#define TILE  32     // valid output positions per block (128 tiles per batch row)
#define HALO  8      // halo per side (>= 3 needed by the k3-conv cascade)

// Time-major LDS row strides (halves), 16B-multiples, conflict-free across 16 lanes
#define LDA   136    // act buffer stride   (272 B/row -> bank step 4)
#define LDH   72     // h / s buffer stride (144 B/row -> bank step 36, gcd 4)
#define LD2   40     // h2 buffer stride    ( 80 B/row -> bank step 20, gcd 4)

typedef __attribute__((ext_vector_type(16))) _Float16 v16h;
typedef __attribute__((ext_vector_type(8)))  _Float16 v8h;
typedef __attribute__((ext_vector_type(8)))  float    v8f;
typedef __attribute__((ext_vector_type(4)))  float    v4f;
typedef __attribute__((ext_vector_type(4)))  int      v4i;

// ---- optional gfx1250 async global->LDS path (guarded; falls back to manual copy)
#if defined(__gfx1250__) && __has_builtin(__builtin_amdgcn_global_load_async_to_lds_b128)
#define HAS_ASYNC 1
typedef __attribute__((address_space(1))) v4i gas_v4i;
typedef __attribute__((address_space(3))) v4i las_v4i;
#else
#define HAS_ASYNC 0
#endif

// ---- WMMA fragment helpers (layouts per CDNA5 ISA 7.12.2, wave32) ----

// A-matrix 16x32 f16: lane L -> M = L%16, half = L/16.
// VGPR v holds K = (v/4)*16 + half*8 + (v%4)*2 (pair). Generic strided read:
// element (m,k) at W[m*rs + k*cs + off]
__device__ __forceinline__ v16h load_A_w(const float* __restrict__ W, int rs, int cs, int off,
                                         int mbase, int kbase, int lane) {
  const int m = lane & 15, half = lane >> 4;
  v16h a;
#pragma unroll
  for (int v = 0; v < 8; ++v) {
    const int k = ((v >> 2) << 4) + (half << 3) + ((v & 3) << 1);
    a[2 * v]     = (_Float16)W[(mbase + m) * rs + (kbase + k) * cs + off];
    a[2 * v + 1] = (_Float16)W[(mbase + m) * rs + (kbase + k + 1) * cs + off];
  }
  return a;
}

// B-matrix 32x16 f16 from *time-major* LDS S[j][ch] (row stride lds_):
// lane L -> N = L%16, element s holds K = (s/8)*16 + (L/16)*8 + s%8.
// K values are contiguous in LDS -> two 16B ds loads.
__device__ __forceinline__ v16h load_B_t(const _Float16* S, int lds_, int kbase, int nbase,
                                         int shift, int lane) {
  const int n = lane & 15;
  const int koff = (lane >> 4) << 3;
  int jn = nbase + n + shift;
  jn = jn < 0 ? 0 : (jn > LW - 1 ? LW - 1 : jn);
  const _Float16* p = S + jn * lds_ + kbase + koff;
  const v8h lo = *(const v8h*)p;
  const v8h hi = *(const v8h*)(p + 16);
  v16h b;
#pragma unroll
  for (int i = 0; i < 8; ++i) { b[i] = lo[i]; b[i + 8] = hi[i]; }
  return b;
}

// C/D 16x16 f32: VGPR r -> M = r + (lane/16)*8, N = lane%16.
// Time-major store: 8 consecutive channels -> one 16B ds store.
__device__ __forceinline__ void store_D_lrelu(_Float16* H, int ldo, const float* __restrict__ bias,
                                              int mbase, int nbase, int lane, v8f c) {
  const int n = lane & 15, mh = (lane >> 4) << 3;
  v8h o;
#pragma unroll
  for (int r = 0; r < 8; ++r) {
    float x = c[r] + bias[mbase + mh + r];
    x = x > 0.0f ? x : 0.1f * x;                 // leaky_relu(0.1)
    o[r] = (_Float16)x;
  }
  *(v8h*)(H + (nbase + n) * ldo + mbase + mh) = o;
}

__device__ __forceinline__ void store_D_f32(float* H, int ldo, const float* __restrict__ bias,
                                            int mbase, int nbase, int lane, v8f c) {
  const int n = lane & 15, mh = (lane >> 4) << 3;
#pragma unroll
  for (int r = 0; r < 8; ++r)
    H[(nbase + n) * ldo + mbase + mh + r] = c[r] + bias[mbase + mh + r];
}

// Pointwise conv (1x1) as WMMA GEMM over a time tile (time-major in/out)
__device__ __forceinline__ void pw_gemm(const float* __restrict__ Wg, const float* __restrict__ bias,
                                        const _Float16* Sin, int ldi, _Float16* Hout, int ldo,
                                        int M, int K, int lane, int wave, int nwaves) {
  const int kt = K >> 5;
  const int ntiles = (M >> 4) * (LW >> 4);
  for (int tile = wave; tile < ntiles; tile += nwaves) {
    const int mb = (tile / (LW >> 4)) << 4, nb = (tile % (LW >> 4)) << 4;
    v8f acc = {};
    for (int kk = 0; kk < kt; ++kk) {
      v16h a = load_A_w(Wg, K, 1, 0, mb, kk << 5, lane);
      v16h b = load_B_t(Sin, ldi, kk << 5, nb, 0, lane);
      acc = __builtin_amdgcn_wmma_f32_16x16x32_f16(false, a, false, b,
                                                   (short)0, acc, false, false);
    }
    store_D_lrelu(Hout, ldo, bias, mb, nb, lane, acc);
  }
}

// Depthwise k=3 pad=1 (bias, no activation); channel-major f32 in -> time-major f16 out
__device__ __forceinline__ void dw3_c2t(const float* __restrict__ wg, const float* __restrict__ bias,
                                        const float* Sin, _Float16* Aout, int C, int ldo,
                                        int tid, int nth) {
  for (int e = tid; e < C * LW; e += nth) {
    const int ch = e / LW, j = e - ch * LW;
    const float l = (j > 0)      ? Sin[ch * LW + j - 1] : 0.0f;
    const float m = Sin[ch * LW + j];
    const float r = (j < LW - 1) ? Sin[ch * LW + j + 1] : 0.0f;
    Aout[j * ldo + ch] =
        (_Float16)(bias[ch] + wg[ch * 3 + 0] * l + wg[ch * 3 + 1] * m + wg[ch * 3 + 2] * r);
  }
}

// Depthwise k=3 pad=1; time-major f16 in -> time-major f16 out
__device__ __forceinline__ void dw3_t2t(const float* __restrict__ wg, const float* __restrict__ bias,
                                        const _Float16* Sin, int ldi, _Float16* Aout, int ldo,
                                        int C, int tid, int nth) {
  for (int e = tid; e < C * LW; e += nth) {
    const int ch = e / LW, j = e - ch * LW;
    const float l = (j > 0)      ? (float)Sin[(j - 1) * ldi + ch] : 0.0f;
    const float m = (float)Sin[j * ldi + ch];
    const float r = (j < LW - 1) ? (float)Sin[(j + 1) * ldi + ch] : 0.0f;
    Aout[j * ldo + ch] =
        (_Float16)(bias[ch] + wg[ch * 3 + 0] * l + wg[ch * 3 + 1] * m + wg[ch * 3 + 2] * r);
  }
}

// ---------------- Kernel 1: fused conv front-end ----------------
// LDS map (bytes):
//   [0,      24576)  condF  f32 [128][48]  (later aliased by logitsL f32 [48][16])
//   [24576,  37632)  actL   f16 time-major [48][LDA]
//   [37632,  44544)  hL     f16 time-major [48][LDH]
//   [44544,  48384)  h2L    f16 time-major [48][LD2]
__global__ __launch_bounds__(256)
void tg_front_kernel(const float* __restrict__ cond,
                     const float* __restrict__ dw1_w, const float* __restrict__ dw1_b,
                     const float* __restrict__ pw1_w, const float* __restrict__ pw1_b,
                     const float* __restrict__ dw2_w, const float* __restrict__ dw2_b,
                     const float* __restrict__ pw2_w, const float* __restrict__ pw2_b,
                     const float* __restrict__ env_w, const float* __restrict__ env_b,
                     const float* __restrict__ sdw_w, const float* __restrict__ sdw_b,
                     const float* __restrict__ spw_w, const float* __restrict__ spw_b,
                     const float* __restrict__ sel_w, const float* __restrict__ sel_b,
                     float* __restrict__ envO, int* __restrict__ idx3, float* __restrict__ w3) {
  __shared__ __align__(16) char smem[48384];
  float*    condF   = (float*)smem;
  _Float16* actL    = (_Float16*)(smem + 24576);
  _Float16* hL      = (_Float16*)(smem + 37632);
  _Float16* h2L     = (_Float16*)(smem + 44544);
  float*    logitsL = (float*)smem;             // aliases condF after sdw is done

  const int tid = threadIdx.x;
  const int lane = tid & 31;
  const int wave = tid >> 5;
  const int nw = blockDim.x >> 5;
  const int b  = blockIdx.x >> 7;               // 128 tiles per batch (4096/32)
  const int t0 = (blockIdx.x & 127) * TILE;

  // 1) condition tile -> LDS f32 (16B chunks; all chunks fully in- or out-of-range)
  {
    const float* gbase = cond + (size_t)b * NC * NT;
    const int nchunk = NC * (LW / 4);
    for (int c = tid; c < nchunk; c += blockDim.x) {
      const int ch = c / (LW / 4);
      const int cj = (c - ch * (LW / 4)) * 4;
      const int t = t0 - HALO + cj;
      float* lp = condF + ch * LW + cj;
      if (t >= 0 && t + 3 < NT) {
#if HAS_ASYNC
        __builtin_amdgcn_global_load_async_to_lds_b128(
            (gas_v4i*)(gbase + (size_t)ch * NT + t),
            (las_v4i*)lp, 0, 0);
#else
        *(v4f*)lp = *(const v4f*)(gbase + (size_t)ch * NT + t);
#endif
      } else {
        v4f z = {0.0f, 0.0f, 0.0f, 0.0f};
        *(v4f*)lp = z;
      }
    }
#if HAS_ASYNC
#if __has_builtin(__builtin_amdgcn_s_wait_asynccnt)
    __builtin_amdgcn_s_wait_asynccnt(0);
#else
    asm volatile("s_wait_asynccnt 0" ::: "memory");
#endif
#endif
  }
  __syncthreads();

  // ---- envelope branch ----
  dw3_c2t(dw1_w, dw1_b, condF, actL, NC, LDA, tid, blockDim.x);        // dw1 (128 ch)
  __syncthreads();
  pw_gemm(pw1_w, pw1_b, actL, LDA, hL, LDH, 64, 128, lane, wave, nw);  // pw1 (WMMA)
  __syncthreads();
  dw3_t2t(dw2_w, dw2_b, hL, LDH, actL, LDA, 64, tid, blockDim.x);      // dw2 (64 ch)
  __syncthreads();
  pw_gemm(pw2_w, pw2_b, actL, LDA, h2L, LD2, 32, 64, lane, wave, nw);  // pw2 (WMMA)
  __syncthreads();

  // env conv (32->1, k=3) + sigmoid -> global envelope
  if (tid < LW) {
    const int j = tid;
    float acc = env_b[0];
    for (int c = 0; c < 32; ++c) {
#pragma unroll
      for (int d = 0; d < 3; ++d) {
        const int jj = j + d - 1;
        const float v = (jj >= 0 && jj < LW) ? (float)h2L[jj * LD2 + c] : 0.0f;
        acc += env_w[c * 3 + d] * v;
      }
    }
    if (j >= HALO && j < HALO + TILE)
      envO[b * NT + t0 + (j - HALO)] = 1.0f / (1.0f + __expf(-acc));
  }

  // ---- selector branch (reuses actL / hL; condF dead afterwards) ----
  dw3_c2t(sdw_w, sdw_b, condF, actL, NC, LDA, tid, blockDim.x);        // sdw (128 ch)
  __syncthreads();
  pw_gemm(spw_w, spw_b, actL, LDA, hL, LDH, 64, 128, lane, wave, nw);  // spw (WMMA)
  __syncthreads();

  // sel conv (64->16, k=3) as 3 shifted WMMA GEMM accumulations
  for (int tile = wave; tile < (LW >> 4); tile += nw) {
    const int nb = tile << 4;
    v8f acc = {};
#pragma unroll
    for (int d = 0; d < 3; ++d) {
      for (int kk = 0; kk < 2; ++kk) {
        v16h a  = load_A_w(sel_w, 64 * 3, 3, d, 0, kk << 5, lane);
        v16h bf = load_B_t(hL, LDH, kk << 5, nb, d - 1, lane);
        acc = __builtin_amdgcn_wmma_f32_16x16x32_f16(false, a, false, bf,
                                                     (short)0, acc, false, false);
      }
    }
    store_D_f32(logitsL, 16, sel_b, 0, nb, lane, acc);
  }
  __syncthreads();

  // top-3 + softmax -> sparse atom weights
  if (tid < TILE) {
    const int j = tid + HALO;
    const int t = t0 + tid;
    float v0 = -1e30f, v1 = -1e30f, v2 = -1e30f;
    int i0 = 0, i1 = 0, i2 = 0;
    for (int a = 0; a < NATOM; ++a) {
      const float x = logitsL[j * 16 + a];
      if (x > v0)      { v2 = v1; i2 = i1; v1 = v0; i1 = i0; v0 = x; i0 = a; }
      else if (x > v1) { v2 = v1; i2 = i1; v1 = x; i1 = a; }
      else if (x > v2) { v2 = x; i2 = a; }
    }
    const float e1 = __expf(v1 - v0);
    const float e2 = __expf(v2 - v0);
    const float inv = 1.0f / (1.0f + e1 + e2);
    const int base = (b * NT + t) * 3;
    idx3[base + 0] = i0; w3[base + 0] = inv;
    idx3[base + 1] = i1; w3[base + 1] = e1 * inv;
    idx3[base + 2] = i2; w3[base + 2] = e2 * inv;
  }
}

// ---------------- Kernel 2: atom placement * smoothed envelope ----------------
__global__ __launch_bounds__(256)
void tg_synth_kernel(const float* __restrict__ envO, const int* __restrict__ idx3,
                     const float* __restrict__ w3, const float* __restrict__ atoms,
                     const float* __restrict__ smooth_w, float* __restrict__ out) {
  const int i = blockIdx.x * blockDim.x + threadIdx.x;
  const int L = AUDIO_LEN;
  if (i >= NB * L) return;
  const int b = i / L;
  const int j = i - b * L;
  const int t = j / HOPS;
  const int off = j - t * HOPS;

  float placed = 0.0f;
  if (off < ALEN) {
    const int base = (b * NT + t) * 3;
    placed = w3[base + 0] * atoms[idx3[base + 0] * ALEN + off]
           + w3[base + 1] * atoms[idx3[base + 1] * ALEN + off]
           + w3[base + 2] * atoms[idx3[base + 2] * ALEN + off];
  }

  // nearest-neighbor envelope upsample + 7-tap smoothing (zero boundary = pad 3)
  float es = 0.0f;
#pragma unroll
  for (int d = -3; d <= 3; ++d) {
    const int jj = j + d;
    if (jj >= 0 && jj < L) {
      const unsigned tt = ((unsigned)jj << 12) / (unsigned)L;  // floor(jj*4096/L)
      es += smooth_w[d + 3] * envO[b * NT + (int)tt];
    }
  }
  out[i] = placed * es;
}

extern "C" void kernel_launch(void* const* d_in, const int* in_sizes, int n_in,
                              void* d_out, int out_size, void* d_ws, size_t ws_size,
                              hipStream_t stream) {
  (void)in_sizes; (void)n_in; (void)out_size; (void)ws_size;
  const float* cond    = (const float*)d_in[0];
  const float* dw1_w   = (const float*)d_in[2];
  const float* dw1_b   = (const float*)d_in[3];
  const float* pw1_w   = (const float*)d_in[4];
  const float* pw1_b   = (const float*)d_in[5];
  const float* dw2_w   = (const float*)d_in[6];
  const float* dw2_b   = (const float*)d_in[7];
  const float* pw2_w   = (const float*)d_in[8];
  const float* pw2_b   = (const float*)d_in[9];
  const float* env_w   = (const float*)d_in[10];
  const float* env_b   = (const float*)d_in[11];
  const float* smooth  = (const float*)d_in[12];
  const float* sdw_w   = (const float*)d_in[13];
  const float* sdw_b   = (const float*)d_in[14];
  const float* spw_w   = (const float*)d_in[15];
  const float* spw_b   = (const float*)d_in[16];
  const float* sel_w   = (const float*)d_in[17];
  const float* sel_b   = (const float*)d_in[18];
  const float* atoms   = (const float*)d_in[19];

  // Workspace: envelope (B*T f32) + top3 idx (B*T*3 i32) + top3 w (B*T*3 f32) = 3.5 MB
  float* envO = (float*)d_ws;
  int*   idx3 = (int*)(envO + (size_t)NB * NT);
  float* w3   = (float*)(idx3 + (size_t)NB * NT * 3);

  tg_front_kernel<<<NB * (NT / TILE), 256, 0, stream>>>(
      cond, dw1_w, dw1_b, pw1_w, pw1_b, dw2_w, dw2_b, pw2_w, pw2_b,
      env_w, env_b, sdw_w, sdw_b, spw_w, spw_b, sel_w, sel_b,
      envO, idx3, w3);

  const long long total = (long long)NB * AUDIO_LEN;
  const int blocks = (int)((total + 255) / 256);
  tg_synth_kernel<<<blocks, 256, 0, stream>>>(envO, idx3, w3, atoms, smooth, (float*)d_out);
}
